// news_encoder_37924561224017
// MI455X (gfx1250) — matrix operand order repaired and checked by reading, not compiled, not run on hardware
//
#include <hip/hip_runtime.h>
#include <hip/hip_bf16.h>

// ---------------------------------------------------------------------------
// News encoder (multi-head self-attn + LayerNorm + additive pooling), MI455X.
// Strategy: one workgroup (8 wave32) per batch item. All GEMMs via
// v_wmma_f32_16x16x32_f16 (f16 A/B, f32 acc). Weights pre-converted to f16,
// transposed [N][Kpad] by a prep kernel so B-fragments are contiguous loads.
// ---------------------------------------------------------------------------

typedef __attribute__((ext_vector_type(16))) _Float16 v16h;
typedef __attribute__((ext_vector_type(8)))  _Float16 v8h;
typedef __attribute__((ext_vector_type(8)))  float    v8f;

#define SEQ   32      // L
#define WD    300     // word dim
#define KP    320     // word dim padded to 32
#define MD    400     // multi dim (heads*head_dim)
#define QD    200     // query dim
#define KP2   416     // multi dim padded to 32 (K of Wa GEMM)
#define NH    20      // heads
#define HD    20      // head dim
#define HDP   32      // padded head dim

// f16 workspace layout (element offsets)
#define WS_WQ  0
#define WS_WK  (MD*KP)           // 128000
#define WS_WV  (2*MD*KP)
#define WS_WA  (3*MD*KP)         // then [QD][KP2]
#define WS_F16_ELEMS (3*MD*KP + QD*KP2)   // 467200 halfs = 934400 bytes

// LDS layout (byte offsets)
#define OFF_X    0                      // [32][320] f16      = 20480
#define OFF_QA   20480                  // q / attn overlay [20][32][32] f16 = 40960
#define OFF_K    61440                  // k  [20][32][32] f16 = 40960
#define OFF_VT   102400                 // v^T[20][32][32] f16 = 40960
#define OFF_CTX  143360                 // ctx [32][400] f32   = 51200
#define OFF_H16  194560                 // h   [32][416] f16   = 26624
#define OFF_WS   221184                 // per-wave scores [8][32][32] f32 = 32768
#define OFF_S    253952                 // s / alpha [32] f32  = 128
#define SMEM_BYTES 254080

union V16 { v16h v; v8h h[2]; };

// A-fragment, 16x32 f16: lane&15 = row; lane>>4 selects K sub-chunk.
// VGPR0-3 hold K = kb + sel*8 .. +7 ; VGPR4-7 hold K = kb + 16 + sel*8 .. +7
__device__ __forceinline__ v16h load_frag_A(const _Float16* t, int stride,
                                            int row, int kb, int lane) {
  const int sel = (lane >> 4) & 1;
  const _Float16* p = t + row * stride + kb + sel * 8;
  V16 u;
  u.h[0] = *(const v8h*)(p);
  u.h[1] = *(const v8h*)(p + 16);
  return u.v;
}

// B-fragment, 32x16 f16 from column-major-by-N storage [N][Kstride]:
// lane&15 = column N; VGPR0-3 hold K = kb + sel*8 .. +7 ; VGPR4-7 hold K = kb + 16 + sel*8 .. +7
__device__ __forceinline__ v16h load_frag_B(const _Float16* t, int stride,
                                            int col, int kb, int lane) {
  const int sel = (lane >> 4) & 1;
  const _Float16* p = t + col * stride + kb + sel * 8;
  V16 u;
  u.h[0] = *(const v8h*)(p);
  u.h[1] = *(const v8h*)(p + 16);
  return u.v;
}

__device__ __forceinline__ v8f wmma_f16(v16h a, v16h b, v8f c) {
  return __builtin_amdgcn_wmma_f32_16x16x32_f16(false, a, false, b,
                                                (short)0, c, false, false);
}

// ---------------------------------------------------------------------------
// Prep: convert fp32 weights -> f16, transposed [N][Kpad], zero K padding.
// ---------------------------------------------------------------------------
__global__ void prep_weights_kernel(const float* __restrict__ Wq,
                                    const float* __restrict__ Wk,
                                    const float* __restrict__ Wv,
                                    const float* __restrict__ Wa,
                                    _Float16* __restrict__ ws) {
  int idx = blockIdx.x * blockDim.x + threadIdx.x;
  const int WSZ = MD * KP;
  if (idx < 3 * WSZ) {
    int m = idx / WSZ, e = idx % WSZ;
    int n = e / KP, kk = e % KP;
    const float* W = (m == 0) ? Wq : ((m == 1) ? Wk : Wv);
    ws[idx] = (kk < WD) ? (_Float16)W[kk * MD + n] : (_Float16)0.f;
  } else if (idx < WS_F16_ELEMS) {
    int e = idx - 3 * WSZ;
    int n = e / KP2, kk = e % KP2;
    ws[idx] = (kk < MD) ? (_Float16)Wa[kk * QD + n] : (_Float16)0.f;
  }
}

// ---------------------------------------------------------------------------
// Main kernel: one block per batch item.
// ---------------------------------------------------------------------------
__global__ __launch_bounds__(256) void news_encoder_kernel(
    const float* __restrict__ x,
    const float* __restrict__ bq, const float* __restrict__ bk,
    const float* __restrict__ bv,
    const float* __restrict__ ln_g, const float* __restrict__ ln_b,
    const float* __restrict__ ba, const float* __restrict__ qv,
    const _Float16* __restrict__ wT,
    float* __restrict__ out) {
  extern __shared__ char smem[];
  _Float16* sX   = (_Float16*)(smem + OFF_X);    // [32][320]
  _Float16* sQ   = (_Float16*)(smem + OFF_QA);   // [20][32][32] (later attn)
  _Float16* sK   = (_Float16*)(smem + OFF_K);    // [20][32][32]
  _Float16* sVT  = (_Float16*)(smem + OFF_VT);   // [20][32][32] (d-major)
  float*    sCtx = (float*)   (smem + OFF_CTX);  // [32][400]
  _Float16* sH16 = (_Float16*)(smem + OFF_H16);  // [32][416]
  float*    sS   = (float*)   (smem + OFF_S);    // [32]

  const int tid  = threadIdx.x;
  const int lane = tid & 31;
  const int wave = tid >> 5;
  const int bb   = blockIdx.x;

  // ---- phase 0: zero padded LDS regions + load x tile as f16 -------------
  {
    uint32_t* z = (uint32_t*)(smem + OFF_QA);
    for (int i = tid; i < (3 * NH * 32 * HDP * 2) / 4; i += 256) z[i] = 0u;
    uint32_t* z2 = (uint32_t*)(smem + OFF_H16);
    for (int i = tid; i < (SEQ * KP2 * 2) / 4; i += 256) z2[i] = 0u;
    if (tid < SEQ) sS[tid] = 0.f;

    const float* xb = x + (size_t)bb * (SEQ * WD);
    for (int i = tid; i < SEQ * KP; i += 256) {
      int r = i / KP, c = i % KP;
      sX[i] = (c < WD) ? (_Float16)xb[r * WD + c] : (_Float16)0.f;
    }
  }
  __syncthreads();

  // ---- phase 1: Q/K/V projections  [32,320]f16 x [320,400]f16 ------------
  // jobs: 3 mats x 25 n-tiles, round-robin over 8 waves.
  {
    const int nloc = lane & 15;
    for (int job = wave; job < 75; job += 8) {
      const int mat = job / 25, nt = job % 25;
      const _Float16* W = wT + (size_t)mat * (MD * KP);
      const int n = nt * 16 + nloc;      // output feature 0..399
      v8f c0 = {}, c1 = {};
      for (int kt = 0; kt < 10; ++kt) {
        const int kb = kt * 32;
        v16h bf = load_frag_B(W, KP, n, kb, lane);
        v16h a0 = load_frag_A(sX, KP, nloc, kb, lane);
        v16h a1 = load_frag_A(sX, KP, nloc + 16, kb, lane);
        c0 = wmma_f16(a0, bf, c0);
        c1 = wmma_f16(a1, bf, c1);
      }
      const float* bias_p = (mat == 0) ? bq : ((mat == 1) ? bk : bv);
      const float bias = bias_p[n];
      const int hh = n / HD, dd = n % HD;
      for (int r = 0; r < 8; ++r) {
        const int i0 = r + 8 * (lane >> 4);
        const float v0 = c0[r] + bias;
        const float v1 = c1[r] + bias;
        if (mat == 0) {
          sQ[(hh * 32 + i0) * HDP + dd]        = (_Float16)v0;
          sQ[(hh * 32 + i0 + 16) * HDP + dd]   = (_Float16)v1;
        } else if (mat == 1) {
          sK[(hh * 32 + i0) * HDP + dd]        = (_Float16)v0;
          sK[(hh * 32 + i0 + 16) * HDP + dd]   = (_Float16)v1;
        } else {  // store v transposed: [h][d][j]
          sVT[(hh * 32 + dd) * HDP + i0]       = (_Float16)v0;
          sVT[(hh * 32 + dd) * HDP + i0 + 16]  = (_Float16)v1;
        }
      }
    }
  }
  __syncthreads();

  // ---- phase 2: per-head scores -> softmax -> ctx (all WMMA) -------------
  {
    float* wsc = (float*)(smem + OFF_WS) + wave * (SEQ * SEQ);
    const float scl = 0.22360679775f;  // 1/sqrt(20)
    const int nloc = lane & 15;
    for (int h = wave; h < NH; h += 8) {
      _Float16* qh = sQ  + h * SEQ * HDP;
      _Float16* kh = sK  + h * SEQ * HDP;
      _Float16* vh = sVT + h * SEQ * HDP;
      // scores = q k^T / sqrt(d)  (K-dim = 32, zero padded)
      for (int mt = 0; mt < 2; ++mt) {
        v16h a = load_frag_A(qh, HDP, nloc + 16 * mt, 0, lane);
        for (int nt = 0; nt < 2; ++nt) {
          v16h bf = load_frag_B(kh, HDP, nt * 16 + nloc, 0, lane);
          v8f c = {};
          c = wmma_f16(a, bf, c);
          const int j = nt * 16 + nloc;
          for (int r = 0; r < 8; ++r)
            wsc[(r + 8 * (lane >> 4) + 16 * mt) * SEQ + j] = c[r] * scl;
        }
      }
      // softmax, one row per lane; write attn f16 over the q slice
      {
        float mx = -1e30f;
        for (int j = 0; j < SEQ; ++j) mx = fmaxf(mx, wsc[lane * SEQ + j]);
        float sum = 0.f;
        for (int j = 0; j < SEQ; ++j) {
          float t = __expf(wsc[lane * SEQ + j] - mx);
          sum += t;
          wsc[lane * SEQ + j] = t;
        }
        const float rs = 1.f / sum;
        for (int j = 0; j < SEQ; ++j)
          qh[lane * HDP + j] = (_Float16)(wsc[lane * SEQ + j] * rs);
      }
      // ctx = attn @ v   (A = attn [i][j], B = v^T [d][j])
      for (int mt = 0; mt < 2; ++mt) {
        v16h a = load_frag_A(qh, HDP, nloc + 16 * mt, 0, lane);
        for (int nt = 0; nt < 2; ++nt) {
          v16h bf = load_frag_B(vh, HDP, nt * 16 + nloc, 0, lane);
          v8f c = {};
          c = wmma_f16(a, bf, c);
          const int dd = nt * 16 + nloc;
          if (dd < HD) {
            const int f = h * HD + dd;
            for (int r = 0; r < 8; ++r)
              sCtx[(r + 8 * (lane >> 4) + 16 * mt) * MD + f] = c[r];
          }
        }
      }
    }
  }
  __syncthreads();

  // ---- phase 3: LayerNorm over 400 features, rows split over waves -------
  for (int i = wave; i < SEQ; i += 8) {
    float s = 0.f, sq = 0.f;
    for (int d = lane; d < MD; d += 32) {
      float v = sCtx[i * MD + d];
      s += v; sq += v * v;
    }
    for (int o = 16; o > 0; o >>= 1) {
      s  += __shfl_xor(s, o, 32);
      sq += __shfl_xor(sq, o, 32);
    }
    const float mu  = s * (1.f / MD);
    const float var = sq * (1.f / MD) - mu * mu;
    const float ri  = rsqrtf(var + 1e-5f);
    for (int d = lane; d < MD; d += 32) {
      float v = (sCtx[i * MD + d] - mu) * ri * ln_g[d] + ln_b[d];
      sH16[i * KP2 + d] = (_Float16)v;   // K padding stays zero
    }
  }
  __syncthreads();

  // ---- phase 4: a = tanh(h @ Wa + ba); s[i] += a*qv (fused, no a buffer) --
  {
    const _Float16* WaT = wT + WS_WA;
    const int nloc = lane & 15;
    for (int nt = wave; nt < 13; nt += 8) {
      const int n = nt * 16 + nloc;
      const bool ok = (n < QD);
      const int ns = ok ? n : 0;
      v8f c0 = {}, c1 = {};
      for (int kt = 0; kt < 13; ++kt) {
        const int kb = kt * 32;
        v16h bf = load_frag_B(WaT, KP2, ns, kb, lane);
        v16h a0 = load_frag_A(sH16, KP2, nloc, kb, lane);
        v16h a1 = load_frag_A(sH16, KP2, nloc + 16, kb, lane);
        c0 = wmma_f16(a0, bf, c0);
        c1 = wmma_f16(a1, bf, c1);
      }
      if (ok) {
        const float bbn = ba[n], qq = qv[n];
        for (int r = 0; r < 8; ++r) {
          const int i0 = r + 8 * (lane >> 4);
          atomicAdd(&sS[i0],      tanhf(c0[r] + bbn) * qq);
          atomicAdd(&sS[i0 + 16], tanhf(c1[r] + bbn) * qq);
        }
      }
    }
  }
  __syncthreads();

  // ---- phase 5: softmax over sequence + weighted pooling -----------------
  if (wave == 0) {
    float mx = -1e30f;
    for (int i = 0; i < SEQ; ++i) mx = fmaxf(mx, sS[i]);
    float sum = 0.f;
    for (int i = 0; i < SEQ; ++i) sum += __expf(sS[i] - mx);
    const float a = __expf(sS[lane] - mx) / sum;
    sS[lane] = a;   // lockstep wave: all reads above precede this store
  }
  __syncthreads();

  float* outb = out + (size_t)bb * MD;
  for (int d = tid; d < MD; d += 256) {
    float acc = 0.f;
    for (int i = 0; i < SEQ; ++i)
      acc += sS[i] * (float)sH16[i * KP2 + d];
    outb[d] = acc;
  }
}

// ---------------------------------------------------------------------------
extern "C" void kernel_launch(void* const* d_in, const int* in_sizes, int n_in,
                              void* d_out, int out_size, void* d_ws, size_t ws_size,
                              hipStream_t stream) {
  const float* x    = (const float*)d_in[0];
  const float* Wq   = (const float*)d_in[1];
  const float* bq   = (const float*)d_in[2];
  const float* Wk   = (const float*)d_in[3];
  const float* bk   = (const float*)d_in[4];
  const float* Wv   = (const float*)d_in[5];
  const float* bv   = (const float*)d_in[6];
  const float* ln_g = (const float*)d_in[7];
  const float* ln_b = (const float*)d_in[8];
  const float* Wa   = (const float*)d_in[9];
  const float* ba   = (const float*)d_in[10];
  const float* qv   = (const float*)d_in[11];

  _Float16* ws = (_Float16*)d_ws;  // needs >= 934,400 bytes
  const int batch = in_sizes[0] / (SEQ * WD);

  prep_weights_kernel<<<(WS_F16_ELEMS + 255) / 256, 256, 0, stream>>>(
      Wq, Wk, Wv, Wa, ws);

  news_encoder_kernel<<<batch, 256, SMEM_BYTES, stream>>>(
      x, bq, bk, bv, ln_g, ln_b, ba, qv, ws, (float*)d_out);
}
